// SpatialAttention_21500606284472
// MI455X (gfx1250) — matrix operand order repaired and checked by reading, not compiled, not run on hardware
//
#include <hip/hip_runtime.h>

// SpatialAttention on MI455X (gfx1250, wave32, WMMA).
//
// Pipeline (all on `stream`, deterministic):
//   0a) x (B,C,HW) f32 -> xT (B,HW,C) bf16        (LDS tile transpose)
//   0b) Wk/Wq/Wv/Wr f32 -> bf16
//   1)  Kt = xT*Wk^T + bk   (B,HW,C)  bf16  [natural store]
//       Qt = xT*Wq^T + bq   (B,HW,C)  bf16  [natural store]
//       V  = (xT*Wv^T + bv)^T (B,C,HW) bf16 [transposed 16B store]
//   2)  scores = Kt * Qt^T  (B,HW,HW) f32
//   3)  softmax over the BATCH axis -> attn (B,HW,HW) bf16   [reuses Qt slab]
//   4)  attT = attn * V^T   (B,HW,C)  bf16   [reuses Kt slab]
//   5)  out = alpha*(attT*Wr^T + br)^T + x   (B,C,HW) f32 -> d_out
//
// GEMM core: 128x128x32 macro-tile, 8 wave32s, each wave owns a 64x32
// sub-tile = 8 x v_wmma_f32_16x16x32_bf16 per K-step (8 v8f accumulators =
// 64 VGPRs -> whole working set stays inside the 256 directly-addressable
// VGPRs: no spills, no s_set_vgpr_msb rotation). Operands are staged into
// double-buffered padded LDS via the CDNA5 async global->LDS path
// (GLOBAL_LOAD_ASYNC_TO_LDS_B128, ASYNCcnt-tracked, bypasses the register
// file) with s_wait_asynccnt pipelining one K-slab ahead.

#define NB   128          // batch
#define CCH  1280         // channels
#define HW   256          // H*W

typedef __attribute__((ext_vector_type(16))) __bf16 v16bf;
typedef __attribute__((ext_vector_type(8)))  float  v8f;

#define BM   128
#define BN   128
#define BK   32
#define LDSK 40           // padded LDS row stride (elements) -> conflict-free frag reads

#if defined(__HIP_DEVICE_COMPILE__) && __has_builtin(__builtin_amdgcn_global_load_async_to_lds_b128)
#define USE_ASYNC 1
#else
#define USE_ASYNC 0
#endif

typedef int v4i __attribute__((vector_size(16)));           // GCC vector: matches builtin pointee
typedef __attribute__((address_space(1))) v4i as1_v4i;      // global ("__device__") int4
typedef __attribute__((address_space(3))) v4i as3_v4i;      // LDS ("__shared__") int4

// 16-byte global -> LDS copy: async (ASYNCcnt) when available, else sync.
__device__ __forceinline__ void cp16(__bf16* dst_lds, const __bf16* src)
{
#if USE_ASYNC
  __builtin_amdgcn_global_load_async_to_lds_b128((as1_v4i*)src, (as3_v4i*)dst_lds, 0, 0);
#else
  *(uint4*)dst_lds = *(const uint4*)src;
#endif
}

template<int N> __device__ __forceinline__ void wait_async()
{
#if USE_ASYNC
#if __has_builtin(__builtin_amdgcn_s_wait_asynccnt)
  __builtin_amdgcn_s_wait_asynccnt(N);
#else
  asm volatile("s_wait_asynccnt %0" :: "i"(N) : "memory");
#endif
#endif
}

union Frag { v16bf v; uint4 q[2]; };

enum { EPI_NAT_BF16 = 0, EPI_TRANS_BF16 = 1, EPI_NAT_F32 = 2, EPI_FINAL = 3 };

template<int M, int NDIM, int K, int EPI>
__global__ __launch_bounds__(256)
void gemm_bf16_wmma(const __bf16* __restrict__ A,  size_t sA,   // (M x K), ld=K, per-batch stride sA
                    const __bf16* __restrict__ Bt, size_t sB,   // (NDIM x K), ld=K, per-batch stride sB
                    void* __restrict__ Dv,         size_t sD,
                    const float* __restrict__ bias,             // len NDIM or null
                    const float* __restrict__ alpha,            // EPI_FINAL only
                    const float* __restrict__ xres)             // EPI_FINAL only
{
  __shared__ __bf16 As[2][BM * LDSK];   // 2 x 10 KB
  __shared__ __bf16 Bs[2][BN * LDSK];   // 2 x 10 KB

  const int tid   = threadIdx.x;
  const int batch = blockIdx.y;
  constexpr int NTN = NDIM / BN;
  const int bm = (int)blockIdx.x / NTN;
  const int bn = (int)blockIdx.x % NTN;

  const __bf16* Ag = A  + (size_t)batch * sA + (size_t)(bm * BM) * K;
  const __bf16* Bg = Bt + (size_t)batch * sB + (size_t)(bn * BN) * K;

  const int wid  = tid >> 5;
  const int lane = tid & 31;
  const int wm   = (wid >> 2) * 64;      // wave M offset within macro-tile (0/64)
  const int wn   = (wid & 3) * 32;       // wave N offset (0/32/64/96)
  const int half = lane >> 4;
  const int lc   = lane & 15;

  v8f acc[4][2] = {};

  const int lr = tid >> 2;               // 0..63  : cooperative-load row
  const int lk = (tid & 3) * 8;          // 0/8/16/24 : 16B chunk within K-slab

  // Issue one K-slab (A: 128x32, B: 128x32) into LDS buffer s. 4 copies/thread.
  auto issue_tile = [&](int k0, int s) {
    const __bf16* Ap = Ag + (size_t)lr * K + k0 + lk;
    cp16(&As[s][lr * LDSK + lk],          Ap);
    cp16(&As[s][(lr + 64) * LDSK + lk],   Ap + (size_t)64 * K);
    const __bf16* Bp = Bg + (size_t)lr * K + k0 + lk;
    cp16(&Bs[s][lr * LDSK + lk],          Bp);
    cp16(&Bs[s][(lr + 64) * LDSK + lk],   Bp + (size_t)64 * K);
  };

  constexpr int NK = K / BK;
  issue_tile(0, 0);

  for (int kt = 0; kt < NK; ++kt) {
    const int cur = kt & 1;
    if (kt + 1 < NK) {
      issue_tile((kt + 1) * BK, cur ^ 1);   // prev compute barrier made buf cur^1 free
      wait_async<4>();                       // tile kt arrived; tile kt+1 may stay in flight
    } else {
      wait_async<0>();
    }
    __syncthreads();                         // all waves' slabs visible

    Frag af[4], bf[2];
#pragma unroll
    for (int i = 0; i < 4; ++i) {            // A frag: lane half -> K {0..7,16..23}/{8..15,24..31}
      const __bf16* p = &As[cur][(wm + i * 16 + lc) * LDSK];
      af[i].q[0] = *(const uint4*)(p + half * 8);
      af[i].q[1] = *(const uint4*)(p + 16 + half * 8);
    }
#pragma unroll
    for (int j = 0; j < 2; ++j) {            // B frag: lane half -> K {0..7,16..23}/{8..15,24..31}
      const __bf16* p = &Bs[cur][(wn + j * 16 + lc) * LDSK];
      bf[j].q[0] = *(const uint4*)(p + half * 8);
      bf[j].q[1] = *(const uint4*)(p + 16 + half * 8);
    }
#pragma unroll
    for (int i = 0; i < 4; ++i)
#pragma unroll
      for (int j = 0; j < 2; ++j)
        acc[i][j] = __builtin_amdgcn_wmma_f32_16x16x32_bf16(
            false, af[i].v, false, bf[j].v, (short)0, acc[i][j], false, false);

    __syncthreads();                         // buf cur free for re-issue at kt+1
  }

  if constexpr (EPI == EPI_NAT_BF16) {
    // D row-major (M x NDIM) bf16, bias over NDIM column
    __bf16* Dp = (__bf16*)Dv + (size_t)batch * sD;
#pragma unroll
    for (int i = 0; i < 4; ++i)
#pragma unroll
      for (int j = 0; j < 2; ++j) {
        const int gm = bm * BM + wm + i * 16;
        const int gn = bn * BN + wn + j * 16 + lc;
        const float bi = bias ? bias[gn] : 0.0f;
#pragma unroll
        for (int r = 0; r < 8; ++r)
          Dp[(size_t)(gm + r + 8 * half) * NDIM + gn] = (__bf16)(acc[i][j][r] + bi);
      }
  } else if constexpr (EPI == EPI_TRANS_BF16) {
    // D^T row-major (NDIM x M) bf16: per-lane 8 contiguous cols -> one 16B store
    __bf16* Dp = (__bf16*)Dv + (size_t)batch * sD;
#pragma unroll
    for (int i = 0; i < 4; ++i)
#pragma unroll
      for (int j = 0; j < 2; ++j) {
        const int gn = bn * BN + wn + j * 16 + lc;           // out row (channel)
        const int gm = bm * BM + wm + i * 16 + 8 * half;     // out col start (pixel)
        const float bi = bias ? bias[gn] : 0.0f;
        alignas(16) __bf16 tmp[8];
#pragma unroll
        for (int r = 0; r < 8; ++r) tmp[r] = (__bf16)(acc[i][j][r] + bi);
        *(uint4*)(Dp + (size_t)gn * M + gm) = *(const uint4*)tmp;
      }
  } else if constexpr (EPI == EPI_NAT_F32) {
    float* Dp = (float*)Dv + (size_t)batch * sD;
#pragma unroll
    for (int i = 0; i < 4; ++i)
#pragma unroll
      for (int j = 0; j < 2; ++j) {
        const int gm = bm * BM + wm + i * 16;
        const int gn = bn * BN + wn + j * 16 + lc;
#pragma unroll
        for (int r = 0; r < 8; ++r)
          Dp[(size_t)(gm + r + 8 * half) * NDIM + gn] = acc[i][j][r];
      }
  } else {  // EPI_FINAL: out = alpha*(D^T + br) + x, f32, 32B stores
    float* Dp = (float*)Dv + (size_t)batch * sD;
    const float* Xp = xres + (size_t)batch * sD;
    const float al = alpha[0];
#pragma unroll
    for (int i = 0; i < 4; ++i)
#pragma unroll
      for (int j = 0; j < 2; ++j) {
        const int c2 = bn * BN + wn + j * 16 + lc;
        const int ms = bm * BM + wm + i * 16 + 8 * half;
        const float bi = bias[c2];
        const float4 x0 = *(const float4*)(Xp + (size_t)c2 * M + ms);
        const float4 x1 = *(const float4*)(Xp + (size_t)c2 * M + ms + 4);
        float4 o0, o1;
        o0.x = al * (acc[i][j][0] + bi) + x0.x;
        o0.y = al * (acc[i][j][1] + bi) + x0.y;
        o0.z = al * (acc[i][j][2] + bi) + x0.z;
        o0.w = al * (acc[i][j][3] + bi) + x0.w;
        o1.x = al * (acc[i][j][4] + bi) + x1.x;
        o1.y = al * (acc[i][j][5] + bi) + x1.y;
        o1.z = al * (acc[i][j][6] + bi) + x1.z;
        o1.w = al * (acc[i][j][7] + bi) + x1.w;
        *(float4*)(Dp + (size_t)c2 * M + ms)     = o0;
        *(float4*)(Dp + (size_t)c2 * M + ms + 4) = o1;
      }
  }
}

// ---- x (B,C,HW) f32 -> xT (B,HW,C) bf16 via 32x32 LDS tile transpose ----
__global__ __launch_bounds__(256)
void transpose_convert_x(const float* __restrict__ x, __bf16* __restrict__ xT)
{
  __shared__ __bf16 t[32][33];
  const int b  = blockIdx.y;
  const int nt = HW / 32;
  const int c0 = (blockIdx.x / nt) * 32;
  const int n0 = (blockIdx.x % nt) * 32;
  const float*  xb  = x  + (size_t)b * CCH * HW;
  __bf16*       xTb = xT + (size_t)b * HW * CCH;
#pragma unroll
  for (int p = 0; p < 4; ++p) {
    const int idx = p * 256 + threadIdx.x;
    const int ci = idx >> 5, nj = idx & 31;
    t[ci][nj] = (__bf16)xb[(size_t)(c0 + ci) * HW + n0 + nj];
  }
  __syncthreads();
#pragma unroll
  for (int p = 0; p < 4; ++p) {
    const int idx = p * 256 + threadIdx.x;
    const int nj = idx >> 5, ci = idx & 31;
    xTb[(size_t)(n0 + nj) * CCH + c0 + ci] = t[ci][nj];
  }
}

__global__ void convert_w(const float* __restrict__ w0, const float* __restrict__ w1,
                          const float* __restrict__ w2, const float* __restrict__ w3,
                          __bf16* __restrict__ out)
{
  const size_t CC = (size_t)CCH * CCH;
  const size_t i  = (size_t)blockIdx.x * blockDim.x + threadIdx.x;
  if (i < CC) {
    out[i]          = (__bf16)w0[i];
    out[CC + i]     = (__bf16)w1[i];
    out[2 * CC + i] = (__bf16)w2[i];
    out[3 * CC + i] = (__bf16)w3[i];
  }
}

// ---- softmax over the batch axis: one thread per (n,m) pair ----
__global__ void softmax_over_batch(const float* __restrict__ s, __bf16* __restrict__ attn)
{
  const int idx = blockIdx.x * blockDim.x + threadIdx.x;     // 0 .. HW*HW-1
  const size_t stride = (size_t)HW * HW;
  float mx = -3.0e38f;
  for (int b = 0; b < NB; ++b) mx = fmaxf(mx, s[(size_t)b * stride + idx]);
  float sum = 0.0f;
  for (int b = 0; b < NB; ++b) {
    const float e = __expf(s[(size_t)b * stride + idx] - mx);
    sum += e;
    attn[(size_t)b * stride + idx] = (__bf16)e;
  }
  const float inv = 1.0f / sum;
  for (int b = 0; b < NB; ++b)
    attn[(size_t)b * stride + idx] = (__bf16)((float)attn[(size_t)b * stride + idx] * inv);
}

extern "C" void kernel_launch(void* const* d_in, const int* in_sizes, int n_in,
                              void* d_out, int out_size, void* d_ws, size_t ws_size,
                              hipStream_t stream)
{
  const float* x     = (const float*)d_in[0];
  const float* Wk    = (const float*)d_in[1];
  const float* bk    = (const float*)d_in[2];
  const float* Wq    = (const float*)d_in[3];
  const float* bq    = (const float*)d_in[4];
  const float* Wv    = (const float*)d_in[5];
  const float* bv    = (const float*)d_in[6];
  const float* Wr    = (const float*)d_in[7];
  const float* br    = (const float*)d_in[8];
  const float* alpha = (const float*)d_in[9];

  char* ws = (char*)d_ws;
  const size_t SZ_XT = (size_t)NB * HW * CCH * 2;      // 80 MB   (B,HW,C) bf16
  const size_t CC2   = (size_t)CCH * CCH * 2;
  const size_t SZ_W  = 4 * CC2;                        // 12.5 MB
  // total ws need: 4*SZ_XT + SZ_W + 32MB scores  ~= 365 MB

  __bf16* xT  = (__bf16*)(ws);
  __bf16* Wkb = (__bf16*)(ws + SZ_XT);
  __bf16* Wqb = (__bf16*)(ws + SZ_XT + CC2);
  __bf16* Wvb = (__bf16*)(ws + SZ_XT + 2 * CC2);
  __bf16* Wrb = (__bf16*)(ws + SZ_XT + 3 * CC2);
  __bf16* Kt  = (__bf16*)(ws + SZ_XT + SZ_W);
  __bf16* Qt  = (__bf16*)(ws + SZ_XT + SZ_W + SZ_XT);
  __bf16* Vn  = (__bf16*)(ws + SZ_XT + SZ_W + 2 * SZ_XT);
  float*  sc  = (float*) (ws + SZ_XT + SZ_W + 3 * SZ_XT);
  __bf16* attn = Qt;   // Qt dead after scores
  __bf16* attT = Kt;   // Kt dead after scores

  const dim3 blk(256);
  const size_t sXT = (size_t)HW * CCH;                 // per-batch elements

  transpose_convert_x<<<dim3((CCH / 32) * (HW / 32), NB), blk, 0, stream>>>(x, xT);
  convert_w<<<((CCH * CCH) + 255) / 256, blk, 0, stream>>>(Wk, Wq, Wv, Wr, Wkb);

  // QKV: M=256(HW) x NDIM=1280(C), K=1280 -> (2 x 10) tiles per batch
  gemm_bf16_wmma<HW, CCH, CCH, EPI_NAT_BF16><<<dim3(20, NB), blk, 0, stream>>>(
      xT, sXT, Wkb, 0, Kt, sXT, bk, nullptr, nullptr);
  gemm_bf16_wmma<HW, CCH, CCH, EPI_NAT_BF16><<<dim3(20, NB), blk, 0, stream>>>(
      xT, sXT, Wqb, 0, Qt, sXT, bq, nullptr, nullptr);
  gemm_bf16_wmma<HW, CCH, CCH, EPI_TRANS_BF16><<<dim3(20, NB), blk, 0, stream>>>(
      xT, sXT, Wvb, 0, Vn, sXT, bv, nullptr, nullptr);

  // scores: 256 x 256, K=1280 -> (2 x 2) tiles per batch
  gemm_bf16_wmma<HW, HW, CCH, EPI_NAT_F32><<<dim3(4, NB), blk, 0, stream>>>(
      Kt, sXT, Qt, sXT, sc, (size_t)HW * HW, nullptr, nullptr, nullptr);

  softmax_over_batch<<<(HW * HW) / 256, blk, 0, stream>>>(sc, attn);

  // att: 256 x 1280, K=256
  gemm_bf16_wmma<HW, CCH, HW, EPI_NAT_BF16><<<dim3(20, NB), blk, 0, stream>>>(
      attn, (size_t)HW * HW, Vn, sXT, attT, sXT, nullptr, nullptr, nullptr);

  // final: 256 x 1280, K=1280, fused bias + alpha + residual
  gemm_bf16_wmma<HW, CCH, CCH, EPI_FINAL><<<dim3(20, NB), blk, 0, stream>>>(
      attT, sXT, Wrb, 0, d_out, (size_t)CCH * HW, br, alpha, x);
}
